// bertaModel_89378269429981
// MI455X (gfx1250) — matrix operand order repaired and checked by reading, not compiled, not run on hardware
//
#include <hip/hip_runtime.h>

typedef __attribute__((ext_vector_type(16))) __bf16 v16bf;
typedef __attribute__((ext_vector_type(8)))  __bf16 v8bf;
typedef __attribute__((ext_vector_type(8)))  float  v8f;
typedef __attribute__((ext_vector_type(4)))  float  v4f;

#define B_   256
#define S_   512
#define H_   1024
#define H3_  3072          // 3*H
#define KT_  (H3_ / 32)    // 96 K-tiles of 32
#define NT_  (H3_ / 16)    // 192 N-tiles of 16
#define NUMS_ 6

// ---------------------------------------------------------------------------
// 1) Find first occurrence of token 0 (start) and token 2 (end) per row.
// ---------------------------------------------------------------------------
__global__ __launch_bounds__(256) void find_markers_k(const int* __restrict__ ids,
                                                      int* __restrict__ se) {
  __shared__ int s0, s2;
  const int b = blockIdx.x;
  if (threadIdx.x == 0) { s0 = S_; s2 = S_; }
  __syncthreads();
  for (int s = threadIdx.x; s < S_; s += blockDim.x) {
    const int id = ids[b * S_ + s];
    if (id == 0) atomicMin(&s0, s);
    if (id == 2) atomicMin(&s2, s);
  }
  __syncthreads();
  if (threadIdx.x == 0) {
    se[b]      = (s0 == S_) ? 0 : s0;   // argmax semantics: 0 if absent
    se[B_ + b] = (s2 == S_) ? 0 : s2;
  }
}

// ---------------------------------------------------------------------------
// 2) Pack w1 (fp32 row-major [K=3072][N=3072]) into bf16 WMMA B-fragments.
//    Block (kt,nt) = 32x16 tile; lane l = ((kl>>3)&1)*16 + nl holds element
//    s = (kl>>4)*8 + (kl&7), i.e. K = 8*(l/16)+s (s<8), 16+8*(l/16)+s-8 (s>=8). Blocks ordered
//    [nt][kt] so the GEMM K-loop walks contiguous 1KB blocks.
// ---------------------------------------------------------------------------
__global__ __launch_bounds__(256) void pack_w1_k(const float* __restrict__ w1,
                                                 __bf16* __restrict__ wp) {
  const size_t idx = (size_t)blockIdx.x * 256 + threadIdx.x;  // over 3072*3072
  const int k = (int)(idx / H3_);
  const int n = (int)(idx % H3_);
  const int kt = k >> 5, nt = n >> 4;
  const int kl = k & 31, nl = n & 15;
  const int half = (kl >> 3) & 1;
  const int lane = half * 16 + nl;
  const int v = ((kl >> 4) * 8 + (kl & 7)) >> 1;
  const int pos = kl & 1;
  const size_t dst = ((size_t)(nt * KT_ + kt)) * 512 + lane * 16 + v * 2 + pos;
  wp[dst] = (__bf16)w1[idx];
}

// ---------------------------------------------------------------------------
// 3) Segment means: one block per batch row, 1024 threads = 4-way S split x
//    256 float4 lanes over H. Non-temporal loads: the 512 MiB stream is
//    single-use; keep L2 for the 18 MB packed weight. LDS reduce over the
//    4 S-chunks, then divide and store bf16 combined [B,3H].
// ---------------------------------------------------------------------------
__global__ __launch_bounds__(1024) void seg_mean_k(const float* __restrict__ tv,
                                                   const int* __restrict__ se,
                                                   __bf16* __restrict__ comb) {
  const int b  = blockIdx.x;
  const int t  = threadIdx.x;          // 0..1023
  const int h4 = t & 255;              // float4 index over H
  const int sc = t >> 8;               // S-chunk 0..3
  const int start = se[b];
  const int end   = se[B_ + b];

  v4f a1 = (v4f)(0.0f);
  v4f a2 = (v4f)(0.0f);
  v4f a3 = (v4f)(0.0f);

  const v4f* row = (const v4f*)(tv + (size_t)b * S_ * H_) + h4;
  const int s_lo = sc * (S_ / 4);
  const int s_hi = s_lo + (S_ / 4);
  #pragma unroll 4
  for (int s = s_lo; s < s_hi; ++s) {
    const v4f v = __builtin_nontemporal_load(&row[(size_t)s * (H_ / 4)]);
    if (s < start)      a1 += v;
    else if (s <= end)  a2 += v;
    else                a3 += v;
  }

  __shared__ v4f sh[3][1024];          // 48 KB of 320 KB LDS
  sh[0][t] = a1; sh[1][t] = a2; sh[2][t] = a3;
  __syncthreads();

  if (sc == 0) {
    #pragma unroll
    for (int j = 1; j < 4; ++j) {
      a1 += sh[0][j * 256 + h4];
      a2 += sh[1][j * 256 + h4];
      a3 += sh[2][j * 256 + h4];
    }
    const int c1 = start;
    const int c2 = (end >= start) ? (end - start + 1) : 0;
    int c3 = (S_ - 1) - end; if (c3 < 0) c3 = 0;
    const float r1 = (c1 > 0) ? 1.0f / (float)c1 : 0.0f;
    const float r2 = (c2 > 0) ? 1.0f / (float)c2 : 0.0f;
    const float r3 = (c3 > 0) ? 1.0f / (float)c3 : 0.0f;
    __bf16* dst = comb + (size_t)b * H3_;
    const int h = h4 * 4;
    dst[h + 0]          = (__bf16)(a1.x * r1);
    dst[h + 1]          = (__bf16)(a1.y * r1);
    dst[h + 2]          = (__bf16)(a1.z * r1);
    dst[h + 3]          = (__bf16)(a1.w * r1);
    dst[H_ + h + 0]     = (__bf16)(a2.x * r2);
    dst[H_ + h + 1]     = (__bf16)(a2.y * r2);
    dst[H_ + h + 2]     = (__bf16)(a2.z * r2);
    dst[H_ + h + 3]     = (__bf16)(a2.w * r2);
    dst[2 * H_ + h + 0] = (__bf16)(a3.x * r3);
    dst[2 * H_ + h + 1] = (__bf16)(a3.y * r3);
    dst[2 * H_ + h + 2] = (__bf16)(a3.z * r3);
    dst[2 * H_ + h + 3] = (__bf16)(a3.w * r3);
  }
}

// ---------------------------------------------------------------------------
// 4) Split-K GEMM via v_wmma_f32_16x16x32_bf16. Each wave: 16(M)x32(N) tile,
//    half of K (48 steps). Explicit two-stage software pipeline with disjoint
//    register sets (no buffer rotation moves -> no WMMA->VALU hazard NOPs).
//    2 x 16 x 96 = 3072 waves -> 384 blocks. Partials (fp32) -> P[kh][M][N].
// ---------------------------------------------------------------------------
__global__ __launch_bounds__(256) void gemm_splitk_k(const __bf16* __restrict__ X,
                                                     const __bf16* __restrict__ Wp,
                                                     float* __restrict__ P) {
  const int lane = threadIdx.x & 31;
  const int wave = threadIdx.x >> 5;
  const int w    = blockIdx.x * 8 + wave;      // 0..3071
  const int kh   = w / (16 * (NT_ / 2));       // K-half
  const int rem  = w % (16 * (NT_ / 2));
  const int mt   = rem / (NT_ / 2);
  const int nt2  = rem % (NT_ / 2);            // pair of 16-wide N tiles
  const int half = lane >> 4;
  const int mrow = mt * 16 + (lane & 15);

  v8f acc0 = (v8f)(0.0f), acc1 = (v8f)(0.0f);

  const __bf16* xrow = X + (size_t)mrow * H3_;
  const size_t bstride = (size_t)KT_ * 512;    // bf16 elems between n-tiles
  const __bf16* wbase = Wp + ((size_t)(nt2 * 2) * KT_) * 512 + lane * 16;

  const int kt0 = kh * (KT_ / 2);
  const int kt1 = kt0 + (KT_ / 2);             // 48 steps (even)

  // Stage 0: kt0
  v8bf  alo0 = *(const v8bf*)(xrow + kt0 * 32 + half * 8);
  v8bf  ahi0 = *(const v8bf*)(xrow + kt0 * 32 + 16 + half * 8);
  v16bf b00  = *(const v16bf*)(wbase + (size_t)kt0 * 512);
  v16bf b01  = *(const v16bf*)(wbase + (size_t)kt0 * 512 + bstride);
  v8bf  alo1, ahi1;
  v16bf b10, b11;

  for (int kt = kt0; kt < kt1 - 2; kt += 2) {
    // Refill stage 1 with kt+1 while stage 0 computes.
    alo1 = *(const v8bf*)(xrow + (kt + 1) * 32 + half * 8);
    ahi1 = *(const v8bf*)(xrow + (kt + 1) * 32 + 16 + half * 8);
    b10  = *(const v16bf*)(wbase + (size_t)(kt + 1) * 512);
    b11  = *(const v16bf*)(wbase + (size_t)(kt + 1) * 512 + bstride);
    {
      const v16bf a = __builtin_shufflevector(alo0, ahi0,
          0, 1, 2, 3, 4, 5, 6, 7, 8, 9, 10, 11, 12, 13, 14, 15);
      acc0 = __builtin_amdgcn_wmma_f32_16x16x32_bf16(false, a, false, b00,
                                                     (short)0, acc0, false, false);
      acc1 = __builtin_amdgcn_wmma_f32_16x16x32_bf16(false, a, false, b01,
                                                     (short)0, acc1, false, false);
    }
    // Refill stage 0 with kt+2 while stage 1 computes.
    alo0 = *(const v8bf*)(xrow + (kt + 2) * 32 + half * 8);
    ahi0 = *(const v8bf*)(xrow + (kt + 2) * 32 + 16 + half * 8);
    b00  = *(const v16bf*)(wbase + (size_t)(kt + 2) * 512);
    b01  = *(const v16bf*)(wbase + (size_t)(kt + 2) * 512 + bstride);
    {
      const v16bf a = __builtin_shufflevector(alo1, ahi1,
          0, 1, 2, 3, 4, 5, 6, 7, 8, 9, 10, 11, 12, 13, 14, 15);
      acc0 = __builtin_amdgcn_wmma_f32_16x16x32_bf16(false, a, false, b10,
                                                     (short)0, acc0, false, false);
      acc1 = __builtin_amdgcn_wmma_f32_16x16x32_bf16(false, a, false, b11,
                                                     (short)0, acc1, false, false);
    }
  }
  // Tail pair: kt1-2 (in stage 0) and kt1-1 (load into stage 1).
  alo1 = *(const v8bf*)(xrow + (kt1 - 1) * 32 + half * 8);
  ahi1 = *(const v8bf*)(xrow + (kt1 - 1) * 32 + 16 + half * 8);
  b10  = *(const v16bf*)(wbase + (size_t)(kt1 - 1) * 512);
  b11  = *(const v16bf*)(wbase + (size_t)(kt1 - 1) * 512 + bstride);
  {
    const v16bf a = __builtin_shufflevector(alo0, ahi0,
        0, 1, 2, 3, 4, 5, 6, 7, 8, 9, 10, 11, 12, 13, 14, 15);
    acc0 = __builtin_amdgcn_wmma_f32_16x16x32_bf16(false, a, false, b00,
                                                   (short)0, acc0, false, false);
    acc1 = __builtin_amdgcn_wmma_f32_16x16x32_bf16(false, a, false, b01,
                                                   (short)0, acc1, false, false);
  }
  {
    const v16bf a = __builtin_shufflevector(alo1, ahi1,
        0, 1, 2, 3, 4, 5, 6, 7, 8, 9, 10, 11, 12, 13, 14, 15);
    acc0 = __builtin_amdgcn_wmma_f32_16x16x32_bf16(false, a, false, b10,
                                                   (short)0, acc0, false, false);
    acc1 = __builtin_amdgcn_wmma_f32_16x16x32_bf16(false, a, false, b11,
                                                   (short)0, acc1, false, false);
  }

  // C/D layout: VGPR r -> M = r + 8*half, N = lane%16.
  const int col = lane & 15;
  float* pbase = P + (size_t)kh * B_ * H3_;
  #pragma unroll
  for (int r = 0; r < 8; ++r) {
    const int m = mt * 16 + half * 8 + r;
    pbase[(size_t)m * H3_ + nt2 * 32 + col]      = acc0[r];
    pbase[(size_t)m * H3_ + nt2 * 32 + 16 + col] = acc1[r];
  }
}

// ---------------------------------------------------------------------------
// 5) Combine split-K partials + bias + tanh -> bf16 activation.
// ---------------------------------------------------------------------------
__global__ __launch_bounds__(256) void bias_tanh_k(const float* __restrict__ P,
                                                   const float* __restrict__ bias,
                                                   __bf16* __restrict__ Y) {
  const size_t idx = (size_t)blockIdx.x * 256 + threadIdx.x;  // over 256*3072
  const int n = (int)(idx % H3_);
  const float v = P[idx] + P[(size_t)B_ * H3_ + idx] + bias[n];
  Y[idx] = (__bf16)tanhf(v);
}

// ---------------------------------------------------------------------------
// 6) Classifier: [256,3072] @ [3072,6] + b_cls, block-per-row reduction.
// ---------------------------------------------------------------------------
__global__ __launch_bounds__(256) void classifier_k(const __bf16* __restrict__ X,
                                                    const float* __restrict__ wc,
                                                    const float* __restrict__ bc,
                                                    float* __restrict__ out) {
  const int b = blockIdx.x;
  const int t = threadIdx.x;
  float p[NUMS_];
  #pragma unroll
  for (int c = 0; c < NUMS_; ++c) p[c] = 0.0f;
  for (int k = t; k < H3_; k += 256) {
    const float x = (float)X[(size_t)b * H3_ + k];
    #pragma unroll
    for (int c = 0; c < NUMS_; ++c) p[c] += x * wc[(size_t)k * NUMS_ + c];
  }
  __shared__ float sh[NUMS_][256];
  #pragma unroll
  for (int c = 0; c < NUMS_; ++c) sh[c][t] = p[c];
  __syncthreads();
  for (int off = 128; off > 0; off >>= 1) {
    if (t < off) {
      #pragma unroll
      for (int c = 0; c < NUMS_; ++c) sh[c][t] += sh[c][t + off];
    }
    __syncthreads();
  }
  if (t < NUMS_) out[b * NUMS_ + t] = sh[t][0] + bc[t];
}

// ---------------------------------------------------------------------------
extern "C" void kernel_launch(void* const* d_in, const int* in_sizes, int n_in,
                              void* d_out, int out_size, void* d_ws, size_t ws_size,
                              hipStream_t stream) {
  const float* tv  = (const float*)d_in[0];   // [256,512,1024] fp32
  const int*   ids = (const int*)d_in[1];     // [256,512] int32
  const float* w1  = (const float*)d_in[2];   // [3072,3072] fp32
  const float* b1  = (const float*)d_in[3];   // [3072] fp32
  const float* wc  = (const float*)d_in[4];   // [3072,6] fp32
  const float* bc  = (const float*)d_in[5];   // [6] fp32
  float* out = (float*)d_out;                 // [256,6] fp32

  char* ws = (char*)d_ws;
  const size_t SZ_ACT = (size_t)B_ * H3_ * sizeof(__bf16);   // 1.5 MB each
  const size_t SZ_W   = (size_t)H3_ * H3_ * sizeof(__bf16);  // 18 MB
  int*    se   = (int*)ws;                                   // 2 KB
  __bf16* comb = (__bf16*)(ws + 4096);
  __bf16* h1   = (__bf16*)(ws + 4096 + SZ_ACT);
  __bf16* h2   = (__bf16*)(ws + 4096 + 2 * SZ_ACT);
  __bf16* wp   = (__bf16*)(ws + 4096 + 3 * SZ_ACT);
  float*  P    = (float*)(ws + 4096 + 3 * SZ_ACT + SZ_W);    // 6 MB split-K partials

  find_markers_k<<<B_, 256, 0, stream>>>(ids, se);
  pack_w1_k<<<(H3_ * H3_) / 256, 256, 0, stream>>>(w1, wp);
  seg_mean_k<<<B_, 1024, 0, stream>>>(tv, se, comb);

  const int gemm_blocks = 2 * 16 * (NT_ / 2) / 8;            // 384
  const int act_blocks  = (B_ * H3_) / 256;                  // 3072

  gemm_splitk_k<<<gemm_blocks, 256, 0, stream>>>(comb, wp, P);
  bias_tanh_k<<<act_blocks, 256, 0, stream>>>(P, b1, h1);
  gemm_splitk_k<<<gemm_blocks, 256, 0, stream>>>(h1, wp, P);
  bias_tanh_k<<<act_blocks, 256, 0, stream>>>(P, b1, h2);

  classifier_k<<<B_, 256, 0, stream>>>(h2, wc, bc, out);
}
